// GritTransformerLayer_39539468927448
// MI455X (gfx1250) — compile-verified, matrix operand order checked by reading, not run on hardware
//
#include <hip/hip_runtime.h>
#include <hip/hip_bf16.h>
#include <math.h>

typedef __attribute__((ext_vector_type(16))) _Float16 v16h;
typedef __attribute__((ext_vector_type(8)))  _Float16 v8h;
typedef __attribute__((ext_vector_type(8)))  float    v8f;
typedef unsigned int u32;
typedef u32 u32x4 __attribute__((ext_vector_type(4)));
typedef u32 u32x8 __attribute__((ext_vector_type(8)));

#define DIM   128
#define HEADS 8
#define HDIM  16

// ---------------------------------------------------------------------------
// float atomic max via CAS (compares as float each round -> sign-safe)
// ---------------------------------------------------------------------------
__device__ inline void atomic_max_f32(float* addr, float val) {
  unsigned int* ai = reinterpret_cast<unsigned int*>(addr);
  unsigned int cur = __float_as_uint(*addr);
  while (__uint_as_float(cur) < val) {
    unsigned int assumed = cur;
    cur = atomicCAS(ai, assumed, __float_as_uint(val));
    if (cur == assumed) break;
  }
}

// ---------------------------------------------------------------------------
// Generic GEMM: C[M,N] = A[M,K] @ W[K,N] (+bias) (opt relu), f32 in/out,
// f16 WMMA accumulate-in-f32. 8 waves / block, one 16x16 tile per wave.
//  - A tile (16 x K f32, contiguous) is DMA'd to LDS by the Tensor Data
//    Mover (wave 0 builds the D#, issues tensor_load_to_lds, drains
//    TENSORcnt after the W stage so DMA overlaps staging).
//  - W tile (K x 128) is cooperatively staged to LDS as f16 column-major
//    with +8-half padding so every B fragment is two aligned ds_load_b128.
// M % 16 == 0, N % 128 == 0 assumed (true for all call sites here).
// ---------------------------------------------------------------------------
template<int K>
__global__ void __launch_bounds__(256)
k_gemm_wmma(const float* __restrict__ A, const float* __restrict__ W,
            const float* __restrict__ bias, float* __restrict__ C,
            int M, int N, int relu)
{
  constexpr int KP = K + 8;                // padded column pitch (halves)
  __shared__ float    sAf[16 * K];         // f32 A tile (TDM destination)
  __shared__ _Float16 sW[128 * KP];        // f16 W tile, column-major
  const int tid  = threadIdx.x;
  const int lane = tid & 31;
  const int wave = tid >> 5;
  const int row0 = blockIdx.x << 4;
  const int cb   = blockIdx.y << 7;        // first of 128 columns this block

  if (wave == 0) {
    // ---- Tensor DMA: 2D tile, data_size=4B, K x 16 rows, contiguous ----
    unsigned lds_off = (unsigned)(size_t)(&sAf[0]);
    unsigned long long ga = (unsigned long long)(size_t)(A + (size_t)row0 * K);
    u32x4 g0;
    g0[0] = 1u;                                            // count=1 (valid D#)
    g0[1] = lds_off;                                       // lds_addr (bytes)
    g0[2] = (u32)ga;                                       // global_addr[31:0]
    g0[3] = (u32)((ga >> 32) & 0x01FFFFFFu) | (2u << 30);  // ga[56:32] | type=2
    u32x8 g1;
    g1[0] = 2u << 16;                                      // data_size = 4 bytes
    g1[1] = ((u32)K & 0xFFFFu) << 16;                      // tensor_dim0[15:0]
    g1[2] = ((u32)K >> 16) | (16u << 16);                  // dim0 hi | tensor_dim1=16
    g1[3] = ((u32)K) << 16;                                // tile_dim0 = K
    g1[4] = 16u;                                           // tile_dim1 = 16
    g1[5] = (u32)K;                                        // tensor_dim0_stride lo
    g1[6] = 0u;
    g1[7] = 0u;
    asm volatile("tensor_load_to_lds %0, %1" :: "s"(g0), "s"(g1) : "memory");
  }

  // ---- cooperative W stage: global row-major f32 -> LDS col-major f16 ----
  for (int idx = tid; idx < K * 128; idx += 256) {
    int k = idx >> 7, c = idx & 127;       // consecutive tid -> consecutive c
    sW[c * KP + k] = (_Float16)W[(size_t)k * N + cb + c];
  }

  if (wave == 0) __builtin_amdgcn_s_wait_tensorcnt(0);
  __syncthreads();

  {
    const int hi  = (lane & 16) ? 8 : 0;   // half-wave K offset
    const int mr  = lane & 15;             // row (A) / col (B) within tile
    const int col = cb + wave * 16 + mr;
    const _Float16* wcol = &sW[(wave * 16 + mr) * KP];
    v8f acc = {};
#pragma unroll
    for (int kt = 0; kt < K; kt += 32) {
      v16h a = {};
#pragma unroll
      for (int j = 0; j < 16; ++j) {
        // 16-bit operand layout: halves 0..7 -> K = kt+hi+j,
        //                        halves 8..15 -> K = kt+hi+16+(j-8)
        a[j] = (_Float16)sAf[mr * K + kt + hi + ((j < 8) ? j : j + 8)];
      }
      v8h blo = *(const v8h*)(wcol + kt + hi);        // 16B aligned
      v8h bhi = *(const v8h*)(wcol + kt + hi + 16);   // 16B aligned
      v16h b = __builtin_shufflevector(blo, bhi, 0, 1, 2, 3, 4, 5, 6, 7,
                                       8, 9, 10, 11, 12, 13, 14, 15);
      acc = __builtin_amdgcn_wmma_f32_16x16x32_f16(false, a, false, b,
                                                   (short)0, acc, false, false);
    }
    const float bv = bias ? bias[col] : 0.0f;
#pragma unroll
    for (int v = 0; v < 8; ++v) {          // lanes0-15: M=v, lanes16-31: M=v+8
      float out = acc[v] + bv;
      if (relu) out = fmaxf(out, 0.0f);
      C[(size_t)(row0 + v + hi) * N + col] = out;
    }
  }
}

// ---------------------------------------------------------------------------
__global__ void k_fill(float* __restrict__ p, float v, size_t n) {
  size_t i = (size_t)blockIdx.x * blockDim.x + threadIdx.x;
  size_t st = (size_t)gridDim.x * blockDim.x;
  for (; i < n; i += st) p[i] = v;
}

// per-(edge,head): s = signed_sqrt((K[src]+Q[dst])*E_w)+E_b, score=clip(s.Aw)
__global__ void k_edge_score(const float* __restrict__ Kb, const float* __restrict__ Qb,
                             const float* __restrict__ Eproj, const int* __restrict__ ei,
                             const float* __restrict__ Aw, float* __restrict__ s_buf,
                             float* __restrict__ score, float* __restrict__ smax, int E)
{
  int t = blockIdx.x * blockDim.x + threadIdx.x;
  if (t >= E * HEADS) return;
  int e = t >> 3, h = t & 7;
  int src = ei[e], dst = ei[E + e];
  const float* kp = Kb + (size_t)src * DIM + h * HDIM;
  const float* qp = Qb + (size_t)dst * DIM + h * HDIM;
  const float* ew = Eproj + (size_t)e * (2 * DIM) + h * (2 * HDIM);
  const float* eb = ew + HDIM;
  float* sp = s_buf + (size_t)e * DIM + h * HDIM;
  float sc = 0.0f;
#pragma unroll
  for (int d = 0; d < HDIM; ++d) {
    float sv = (kp[d] + qp[d]) * ew[d];
    float tv = (sv >= 0.0f ? sqrtf(sv) : -sqrtf(-sv)) + eb[d];
    sp[d] = tv;
    sc += tv * Aw[d * HEADS + h];
  }
  sc = fminf(fmaxf(sc, -5.0f), 5.0f);
  score[t] = sc;
  atomic_max_f32(&smax[(size_t)dst * HEADS + h], sc);
}

__global__ void k_edge_exp(const int* __restrict__ ei, const float* __restrict__ smax,
                           float* __restrict__ score, float* __restrict__ ssum,
                           float* __restrict__ deg, int E)
{
  int t = blockIdx.x * blockDim.x + threadIdx.x;
  if (t >= E * HEADS) return;
  int e = t >> 3, h = t & 7;
  int dst = ei[E + e];
  float p = expf(score[t] - smax[(size_t)dst * HEADS + h]);
  score[t] = p;
  atomicAdd(&ssum[(size_t)dst * HEADS + h], p);
  if (h == 0) atomicAdd(&deg[dst], 1.0f);
}

__global__ void k_edge_agg(const int* __restrict__ ei, const float* __restrict__ score,
                           const float* __restrict__ ssum, const float* __restrict__ Vb,
                           const float* __restrict__ s_buf, float* __restrict__ wV,
                           float* __restrict__ rowV, int E)
{
  int t = blockIdx.x * blockDim.x + threadIdx.x;
  if (t >= E * HEADS) return;
  int e = t >> 3, h = t & 7;
  int src = ei[e], dst = ei[E + e];
  float w = score[t] / (ssum[(size_t)dst * HEADS + h] + 1e-16f);
  const float* vp = Vb + (size_t)src * DIM + h * HDIM;
  const float* sp = s_buf + (size_t)e * DIM + h * HDIM;
  float* wvp = wV + (size_t)dst * DIM + h * HDIM;
  float* rvp = rowV + (size_t)dst * DIM + h * HDIM;
#pragma unroll
  for (int d = 0; d < HDIM; ++d) {
    atomicAdd(&wvp[d], vp[d] * w);
    atomicAdd(&rvp[d], sp[d] * w);
  }
}

// wV += rowV @ VeRow (per head), then degree-scaler mix -> hbuf
__global__ void k_node_combine(const float* __restrict__ wV, const float* __restrict__ rowV,
                               const float* __restrict__ VeRow, const float* __restrict__ deg,
                               const float* __restrict__ deg_coef, float* __restrict__ hbuf,
                               int Nn)
{
  int t = blockIdx.x * blockDim.x + threadIdx.x;
  if (t >= Nn * DIM) return;
  int n = t >> 7, j = t & 127;
  int h = j >> 4, c = j & 15;
  float acc = wV[(size_t)t];
  const float* rv = rowV + (size_t)n * DIM + h * HDIM;
#pragma unroll
  for (int d = 0; d < HDIM; ++d)
    acc += rv[d] * VeRow[d * DIM + h * HDIM + c];   // VeRow[d][h][c]
  float ld = log1pf(deg[n]);
  hbuf[(size_t)t] = acc * deg_coef[j * 2 + 0] + acc * ld * deg_coef[j * 2 + 1];
}

// batchnorm over rows: column sums via block-partial + atomic (128 cols)
__global__ void k_bn_accum(const float* __restrict__ a, const float* __restrict__ b,
                           float* __restrict__ sum, float* __restrict__ sumsq, int rows)
{
  int col = threadIdx.x & 127;
  int r0  = blockIdx.x * 2 + (threadIdx.x >> 7);
  float s = 0.0f, s2 = 0.0f;
  for (int r = r0; r < rows; r += gridDim.x * 2) {
    size_t idx = (size_t)r * DIM + col;
    float v = a[idx] + (b ? b[idx] : 0.0f);
    s += v; s2 += v * v;
  }
  atomicAdd(&sum[col], s);
  atomicAdd(&sumsq[col], s2);
}

__global__ void k_bn_apply(const float* __restrict__ a, const float* __restrict__ b,
                           const float* __restrict__ sum, const float* __restrict__ sumsq,
                           const float* __restrict__ g, const float* __restrict__ bet,
                           float* __restrict__ out, int rows)
{
  size_t t = (size_t)blockIdx.x * blockDim.x + threadIdx.x;
  size_t n = (size_t)rows * DIM;
  if (t >= n) return;
  int col = (int)(t & 127);
  float m   = sum[col] / rows;
  float var = sumsq[col] / rows - m * m;
  float v = a[t] + (b ? b[t] : 0.0f);
  out[t] = (v - m) * rsqrtf(var + 1e-5f) * g[col] + bet[col];
}

// ---------------------------------------------------------------------------
static inline void gemm(const float* A, const float* W, const float* bias, float* C,
                        int M, int K, int N, int relu, hipStream_t s) {
  dim3 grid(M / 16, N / 128);
  if (K == 128)
    k_gemm_wmma<128><<<grid, dim3(256), 0, s>>>(A, W, bias, C, M, N, relu);
  else
    k_gemm_wmma<256><<<grid, dim3(256), 0, s>>>(A, W, bias, C, M, N, relu);
}
static inline void fill(float* p, float v, size_t n, hipStream_t s) {
  int blocks = (int)((n + 255) / 256); if (blocks > 4096) blocks = 4096;
  k_fill<<<blocks, 256, 0, s>>>(p, v, n);
}

extern "C" void kernel_launch(void* const* d_in, const int* in_sizes, int n_in,
                              void* d_out, int out_size, void* d_ws, size_t ws_size,
                              hipStream_t stream) {
  const float* x         = (const float*)d_in[0];
  const float* edge_attr = (const float*)d_in[1];
  const int*   ei        = (const int*)  d_in[2];
  const float* Wq  = (const float*)d_in[3];  const float* bq  = (const float*)d_in[4];
  const float* Wk  = (const float*)d_in[5];  const float* Wv  = (const float*)d_in[6];
  const float* We  = (const float*)d_in[7];  const float* be  = (const float*)d_in[8];
  const float* Aw  = (const float*)d_in[9];  const float* VeR = (const float*)d_in[10];
  const float* dcf = (const float*)d_in[11];
  const float* Woh = (const float*)d_in[12]; const float* boh = (const float*)d_in[13];
  const float* Woe = (const float*)d_in[14]; const float* boe = (const float*)d_in[15];
  const float* g1h = (const float*)d_in[16]; const float* b1h = (const float*)d_in[17];
  const float* g1e = (const float*)d_in[18]; const float* b1e = (const float*)d_in[19];
  const float* W1  = (const float*)d_in[20]; const float* b1v = (const float*)d_in[21];
  const float* W2  = (const float*)d_in[22]; const float* b2v = (const float*)d_in[23];
  const float* g2h = (const float*)d_in[24]; const float* b2h = (const float*)d_in[25];

  const int Nn = in_sizes[0] / DIM;     // 50000
  const int E  = in_sizes[1] / DIM;     // 500000
  const size_t NnD = (size_t)Nn * DIM;
  const size_t ED  = (size_t)E  * DIM;

  float* ws = (float*)d_ws;
  size_t off = 0;
  auto take = [&](size_t n) { float* p = ws + off; off += n; return p; };
  float* Qb    = take(NnD);
  float* Kb    = take(NnD);
  float* Vb    = take(NnD);
  float* wV    = take(NnD);
  float* rowV  = take(NnD);
  float* hbuf  = take(NnD);
  float* hproj = take(NnD);
  float* h1    = take(NnD);
  float* h2    = take(NnD);
  float* s_buf = take(ED);
  float* score = take((size_t)E * HEADS);
  float* smax  = take((size_t)Nn * HEADS);
  float* ssum  = take((size_t)Nn * HEADS);
  float* deg   = take((size_t)Nn);
  float* bnS   = take(128);
  float* bnQ   = take(128);
  float* Eproj = take((size_t)E * 2 * DIM);   // reused below once consumed
  float* e_tmp = Eproj;                       // E x 128  (after k_edge_score)
  float* mid   = Eproj + ED;                  // Nn x 256 (after k_edge_score)

  float* out_h = (float*)d_out;               // Nn x 128
  float* out_e = out_h + NnD;                 // E  x 128

  const int EH_blocks = (E * HEADS + 255) / 256;

  // ---- init accumulators ----
  fill(smax, -3.0e38f, (size_t)Nn * HEADS, stream);
  fill(ssum, 0.0f, (size_t)Nn * HEADS, stream);
  fill(deg,  0.0f, (size_t)Nn, stream);
  fill(wV,   0.0f, NnD, stream);
  fill(rowV, 0.0f, NnD, stream);

  // ---- projections (WMMA GEMMs) ----
  gemm(x, Wq, bq, Qb, Nn, DIM, DIM, 0, stream);
  gemm(x, Wk, nullptr, Kb, Nn, DIM, DIM, 0, stream);
  gemm(x, Wv, nullptr, Vb, Nn, DIM, DIM, 0, stream);
  gemm(edge_attr, We, be, Eproj, E, DIM, 2 * DIM, 0, stream);

  // ---- edge messages + segment softmax + aggregation ----
  k_edge_score<<<EH_blocks, 256, 0, stream>>>(Kb, Qb, Eproj, ei, Aw, s_buf, score, smax, E);
  k_edge_exp  <<<EH_blocks, 256, 0, stream>>>(ei, smax, score, ssum, deg, E);
  k_edge_agg  <<<EH_blocks, 256, 0, stream>>>(ei, score, ssum, Vb, s_buf, wV, rowV, E);

  // ---- node update: head mix + degree scaler -> out-proj -> BN1h ----
  k_node_combine<<<(Nn * DIM + 255) / 256, 256, 0, stream>>>(wV, rowV, VeR, deg, dcf, hbuf, Nn);
  gemm(hbuf, Woh, boh, hproj, Nn, DIM, DIM, 0, stream);
  fill(bnS, 0.0f, 256, stream);
  k_bn_accum<<<1024, 256, 0, stream>>>(x, hproj, bnS, bnQ, Nn);
  k_bn_apply<<<(int)((NnD + 255) / 256), 256, 0, stream>>>(x, hproj, bnS, bnQ, g1h, b1h, h1, Nn);

  // ---- FFN + BN2h ----
  gemm(h1, W1, b1v, mid, Nn, DIM, 2 * DIM, 1, stream);
  gemm(mid, W2, b2v, h2, Nn, 2 * DIM, DIM, 0, stream);
  fill(bnS, 0.0f, 256, stream);
  k_bn_accum<<<1024, 256, 0, stream>>>(h1, h2, bnS, bnQ, Nn);
  k_bn_apply<<<(int)((NnD + 255) / 256), 256, 0, stream>>>(h1, h2, bnS, bnQ, g2h, b2h, out_h, Nn);

  // ---- edge output: wE @ Wo_e -> BN1e ----
  gemm(s_buf, Woe, boe, e_tmp, E, DIM, DIM, 0, stream);
  fill(bnS, 0.0f, 256, stream);
  k_bn_accum<<<2048, 256, 0, stream>>>(edge_attr, e_tmp, bnS, bnQ, E);
  k_bn_apply<<<(int)((ED + 255) / 256), 256, 0, stream>>>(edge_attr, e_tmp, bnS, bnQ, g1e, b1e, out_e, E);
}
